// PtnMultiheadAttentionCustomDef_84954453115184
// MI455X (gfx1250) — matrix operand order repaired and checked by reading, not compiled, not run on hardware
//
#include <hip/hip_runtime.h>
#include <math.h>

// ---------------------------------------------------------------------------
// Types for CDNA5 WMMA (wave32)
// ---------------------------------------------------------------------------
typedef __attribute__((ext_vector_type(16))) __bf16        v16bf;
typedef __attribute__((ext_vector_type(8)))  float         v8f;
typedef __attribute__((ext_vector_type(4)))  unsigned int  v4u;

// TDM descriptor group types (probe-confirmed builtin signature, ROCm 7.2)
typedef __attribute__((ext_vector_type(4))) unsigned int tdm4u;
typedef __attribute__((ext_vector_type(8))) int          tdm8i;
typedef __attribute__((ext_vector_type(4))) int          tdm4i;

union Frag {          // 16 bf16 = 32 bytes = two 16B chunks
    v16bf v;
    v4u   q[2];
};

typedef unsigned short u16;

static __device__ __forceinline__ u16 f2bf(float f) {
    unsigned int u = __float_as_uint(f);
    u = (u + 0x7FFFu + ((u >> 16) & 1u)) >> 16;   // round-to-nearest-even
    return (u16)u;
}

static __device__ __forceinline__ v8f zero8() {
    v8f v;
#pragma unroll
    for (int i = 0; i < 8; ++i) v[i] = 0.f;
    return v;
}

static __device__ __forceinline__ v8f wmma_bf16(const v16bf& a, const v16bf& b, const v8f& c) {
    return __builtin_amdgcn_wmma_f32_16x16x32_bf16(false, a, false, b, (short)0, c, false, false);
}

// ---------------------------------------------------------------------------
// Tensor Data Mover: 2D tile load (bf16 elements) global -> LDS.
// Builds D# groups per CDNA5 ISA:
//   g0: count=1 | lds_addr | global_addr[56:0] | type=2
//   g1: data_size=1(2B), tensor_dim0/1, tile_dim0/1, tensor_dim0_stride
// tile rows are packed contiguously at lds_addr (row-major, no padding).
// Issue from ONE wave per workgroup; wait with s_wait_tensorcnt.
// ---------------------------------------------------------------------------
static __device__ __forceinline__ void tdm_load_2d(unsigned lds_addr,
                                                   const void* gaddr,
                                                   unsigned tensor_d0, unsigned tensor_d1,
                                                   unsigned tile_d0, unsigned tile_d1,
                                                   unsigned stride0_elems) {
    unsigned long long ga = (unsigned long long)(uintptr_t)gaddr;
    tdm4u g0;
    g0[0] = 1u;                                            // count=1, user descriptor
    g0[1] = lds_addr;                                      // LDS byte address
    g0[2] = (unsigned)(ga & 0xffffffffu);                  // global_addr[31:0]
    g0[3] = (unsigned)((ga >> 32) & 0x1ffffffu) | 0x80000000u; // addr[56:32] | type=2

    tdm8i g1;
    g1[0] = (int)(1u << 16);                               // data_size=1 (2 bytes)
    g1[1] = (int)((tensor_d0 & 0xffffu) << 16);            // tensor_dim0[15:0]
    g1[2] = (int)(((tensor_d0 >> 16) & 0xffffu) |
                  ((tensor_d1 & 0xffffu) << 16));          // dim0 hi | dim1 lo
    g1[3] = (int)(((tensor_d1 >> 16) & 0xffffu) |
                  ((tile_d0 & 0xffffu) << 16));            // dim1 hi | tile_dim0
    g1[4] = (int)(tile_d1 & 0xffffu);                      // tile_dim1 (tile_dim2=0)
    g1[5] = (int)stride0_elems;                            // tensor_dim0_stride[31:0]
    g1[6] = 0;
    g1[7] = 0;

    tdm4i g2 = {0, 0, 0, 0};
    tdm4i g3 = {0, 0, 0, 0};
#if __has_include(<hip/amd_detail/amd_gfx1250_TDM.h>)
    tdm8i g4 = {0, 0, 0, 0, 0, 0, 0, 0};
    __builtin_amdgcn_tensor_load_to_lds(g0, g1, g2, g3, g4, 0);
#else
    __builtin_amdgcn_tensor_load_to_lds(g0, g1, g2, g3, 0);
#endif
}

// ---------------------------------------------------------------------------
// Problem constants
// ---------------------------------------------------------------------------
#define PB 4
#define PS 2048
#define PE 1024
#define PH 16
#define PD 64

// ---------------------------------------------------------------------------
// fp32 -> bf16 conversion (grid-stride)
// ---------------------------------------------------------------------------
__global__ __launch_bounds__(256) void f2bf_kernel(const float* __restrict__ src,
                                                   u16* __restrict__ dst, int n) {
    int i = blockIdx.x * blockDim.x + threadIdx.x;
    int stride = gridDim.x * blockDim.x;
    for (; i < n; i += stride) dst[i] = f2bf(src[i]);
}

// ---------------------------------------------------------------------------
// GEMM: C[M,N] = A[M,K](bf16) @ W[K,N](bf16) + bias
// Block tile 128x128, 256 threads = 8 waves, wave tile 32x64 (2x4 WMMA).
// A tile staged via TDM (wave 0), W tile staged transposed by all threads.
// TO_QKV=true : write bf16 into head-split [B,H,S,D]
// TO_QKV=false: write fp32 row-major [M,N]
// ---------------------------------------------------------------------------
template <bool TO_QKV>
__global__ __launch_bounds__(256) void gemm_bf16(const u16* __restrict__ A,
                                                 const u16* __restrict__ W,
                                                 const float* __restrict__ bias,
                                                 void* __restrict__ Cout,
                                                 int M, int N, int K) {
    __shared__ __align__(16) u16 As[128 * 32];   // A tile row-major [row][k] (TDM dest)
    __shared__ __align__(16) u16 Bs[128 * 32];   // W tile transposed [n][k]

    const int tid   = threadIdx.x;
    const int lane  = tid & 31;
    const int wid   = tid >> 5;
    const int wm    = wid >> 1;       // 0..3 (M)
    const int wn    = wid & 1;        // 0..1 (N)
    const int lrow  = lane & 15;
    const int khalf = lane >> 4;      // 0/1

    const int m0 = blockIdx.x * 128;
    const int n0 = blockIdx.y * 128;

    const unsigned asBase = (unsigned)(uintptr_t)&As[0];

    v8f acc[2][4];
#pragma unroll
    for (int i = 0; i < 2; ++i)
#pragma unroll
        for (int j = 0; j < 4; ++j) acc[i][j] = zero8();

    const int bkr = tid >> 3, bnc = (tid & 7) * 16;   // W loader coords

    for (int k0 = 0; k0 < K; k0 += 32) {
        // A tile (128x32) via Tensor Data Mover, issued by wave 0 only
        if (wid == 0)
            tdm_load_2d(asBase, &A[(size_t)m0 * K + k0],
                        /*tensor*/ (unsigned)K, 128u,
                        /*tile  */ 32u, 128u,
                        /*stride*/ (unsigned)K);

        if (k0 + 32 < K)   // global_prefetch_b8 on next W tile
            __builtin_prefetch(&W[(size_t)(k0 + 32 + bkr) * N + n0 + bnc], 0, 1);

        // stage W tile transposed: Bs[n][k]  (128 n x 32 k)
        {
            const v4u* src = (const v4u*)&W[(size_t)(k0 + bkr) * N + n0 + bnc];
            v4u w0 = src[0], w1 = src[1];
#pragma unroll
            for (int j = 0; j < 4; ++j) {
                unsigned int u = w0[j];
                Bs[(bnc + 2 * j) * 32 + bkr]     = (u16)(u & 0xffffu);
                Bs[(bnc + 2 * j + 1) * 32 + bkr] = (u16)(u >> 16);
                u = w1[j];
                Bs[(bnc + 8 + 2 * j) * 32 + bkr]     = (u16)(u & 0xffffu);
                Bs[(bnc + 8 + 2 * j + 1) * 32 + bkr] = (u16)(u >> 16);
            }
        }
        if (wid == 0) __builtin_amdgcn_s_wait_tensorcnt(0);
        __syncthreads();

        Frag aF[2], bF[4];
#pragma unroll
        for (int mi = 0; mi < 2; ++mi) {
            const int base = (wm * 32 + mi * 16 + lrow) * 32;
            aF[mi].q[0] = *(const v4u*)&As[base + khalf * 8];
            aF[mi].q[1] = *(const v4u*)&As[base + 16 + khalf * 8];
        }
#pragma unroll
        for (int ni = 0; ni < 4; ++ni) {
            const int base = (wn * 64 + ni * 16 + lrow) * 32 + khalf * 8;
            bF[ni].q[0] = *(const v4u*)&Bs[base];
            bF[ni].q[1] = *(const v4u*)&Bs[base + 16];
        }
#pragma unroll
        for (int mi = 0; mi < 2; ++mi)
#pragma unroll
            for (int ni = 0; ni < 4; ++ni)
                acc[mi][ni] = wmma_bf16(aF[mi].v, bF[ni].v, acc[mi][ni]);
        __syncthreads();
    }

    // epilogue
#pragma unroll
    for (int mi = 0; mi < 2; ++mi) {
#pragma unroll
        for (int ni = 0; ni < 4; ++ni) {
            const int col = n0 + wn * 64 + ni * 16 + lrow;
            const float bv = bias[col];
#pragma unroll
            for (int r = 0; r < 8; ++r) {
                const int row = m0 + wm * 32 + mi * 16 + r + khalf * 8;
                const float val = acc[mi][ni][r] + bv;
                if (TO_QKV) {
                    // [b, s, (h d)] -> [b*H+h][s][d]
                    const int b = row >> 11, s = row & (PS - 1);
                    const int h = col >> 6, d = col & (PD - 1);
                    u16* o = (u16*)Cout;
                    o[(((size_t)b * PH + h) * PS + s) * PD + d] = f2bf(val);
                } else {
                    float* o = (float*)Cout;
                    o[(size_t)row * N + col] = val;
                }
            }
        }
    }
}

// ---------------------------------------------------------------------------
// Flash attention: one block per (b*h, 128-row query slab); 8 waves,
// each wave owns 16 query rows. K tile via TDM; V tile transposed manually;
// online softmax in registers, P re-layout through LDS.
// ---------------------------------------------------------------------------
__global__ __launch_bounds__(256) void flash_attn(const u16* __restrict__ Q,
                                                  const u16* __restrict__ Kb,
                                                  const u16* __restrict__ Vb,
                                                  const int* __restrict__ mask,
                                                  u16* __restrict__ attn) {
    __shared__ __align__(16) u16 Ks[64 * 64];       // K tile row-major [key][d] (TDM dest)
    __shared__ __align__(16) u16 VTs[64 * 64];      // V tile transposed [d][key]
    __shared__ __align__(16) u16 Ps[8 * 16 * 64];   // per-wave P staging

    const int tid   = threadIdx.x;
    const int lane  = tid & 31;
    const int wid   = tid >> 5;
    const int lrow  = lane & 15;
    const int khalf = lane >> 4;

    const int bh = blockIdx.x;
    const int b  = bh >> 4;
    const int h  = bh & 15;
    const int q0 = blockIdx.y * 128 + wid * 16;

    const u16* Qg = Q + (size_t)bh * PS * PD;
    const u16* Kg = Kb + (size_t)bh * PS * PD;
    const u16* Vg = Vb + (size_t)bh * PS * PD;

    const unsigned ksBase = (unsigned)(uintptr_t)&Ks[0];

    // Q fragments for this wave's 16 rows (A-frag layout, direct global loads)
    Frag Qa[2];
    {
        const u16* qp = Qg + (size_t)(q0 + lrow) * PD;
#pragma unroll
        for (int kk = 0; kk < 2; ++kk) {
            Qa[kk].q[0] = *(const v4u*)(qp + kk * 32 + khalf * 8);
            Qa[kk].q[1] = *(const v4u*)(qp + kk * 32 + 16 + khalf * 8);
        }
    }

    v8f O[4];
#pragma unroll
    for (int t = 0; t < 4; ++t) O[t] = zero8();
    float mrow[8], lsum[8];
#pragma unroll
    for (int r = 0; r < 8; ++r) { mrow[r] = -1e30f; lsum[r] = 0.f; }

    const int vkey = tid >> 2, vd0 = (tid & 3) * 16;   // V loader coords

    for (int kb = 0; kb < PS; kb += 64) {
        // K tile via Tensor Data Mover (wave 0 only)
        if (wid == 0)
            tdm_load_2d(ksBase, Kg + (size_t)kb * PD,
                        /*tensor*/ 64u, (unsigned)PS,
                        /*tile  */ 64u, 64u,
                        /*stride*/ 64u);

        // stage V tile transposed (all threads)
        {
            const v4u* vs = (const v4u*)(Vg + (size_t)(kb + vkey) * PD + vd0);
            v4u v0 = vs[0], v1 = vs[1];
#pragma unroll
            for (int j = 0; j < 4; ++j) {
                unsigned int u = v0[j];
                VTs[(vd0 + 2 * j) * 64 + vkey]     = (u16)(u & 0xffffu);
                VTs[(vd0 + 2 * j + 1) * 64 + vkey] = (u16)(u >> 16);
                u = v1[j];
                VTs[(vd0 + 8 + 2 * j) * 64 + vkey]     = (u16)(u & 0xffffu);
                VTs[(vd0 + 8 + 2 * j + 1) * 64 + vkey] = (u16)(u >> 16);
            }
        }
        if (wid == 0) __builtin_amdgcn_s_wait_tensorcnt(0);
        __syncthreads();

        // S = Q K^T  (16 x 64 scores per wave)
        v8f sAcc[4];
#pragma unroll
        for (int t = 0; t < 4; ++t) sAcc[t] = zero8();
#pragma unroll
        for (int kk = 0; kk < 2; ++kk) {
#pragma unroll
            for (int t = 0; t < 4; ++t) {
                Frag bf;
                const int base = (t * 16 + lrow) * 64 + kk * 32 + khalf * 8;
                bf.q[0] = *(const v4u*)&Ks[base];
                bf.q[1] = *(const v4u*)&Ks[base + 16];
                sAcc[t] = wmma_bf16(Qa[kk].v, bf.v, sAcc[t]);
            }
        }

        // scale + key-padding mask
#pragma unroll
        for (int t = 0; t < 4; ++t) {
            const int key = kb + t * 16 + lrow;
            const bool valid = mask[b * PS + key] != 0;
#pragma unroll
            for (int r = 0; r < 8; ++r) {
                const float x = sAcc[t][r] * 0.125f;     // 1/sqrt(64)
                sAcc[t][r] = valid ? x : -__builtin_inff();
            }
        }

        // online softmax (reductions across 16-lane halves = 16 columns each)
        float pv[4][8];
#pragma unroll
        for (int r = 0; r < 8; ++r) {
            float bm = sAcc[0][r];
#pragma unroll
            for (int t = 1; t < 4; ++t) bm = fmaxf(bm, sAcc[t][r]);
            bm = fmaxf(bm, __shfl_xor(bm, 1, 32));
            bm = fmaxf(bm, __shfl_xor(bm, 2, 32));
            bm = fmaxf(bm, __shfl_xor(bm, 4, 32));
            bm = fmaxf(bm, __shfl_xor(bm, 8, 32));
            const float mn = fmaxf(mrow[r], bm);
            const float sc = __expf(mrow[r] - mn);
            float rs = 0.f;
#pragma unroll
            for (int t = 0; t < 4; ++t) {
                const float p = __expf(sAcc[t][r] - mn);
                pv[t][r] = p;
                rs += p;
            }
            rs += __shfl_xor(rs, 1, 32);
            rs += __shfl_xor(rs, 2, 32);
            rs += __shfl_xor(rs, 4, 32);
            rs += __shfl_xor(rs, 8, 32);
            lsum[r] = lsum[r] * sc + rs;
            mrow[r] = mn;
#pragma unroll
            for (int t = 0; t < 4; ++t) O[t][r] *= sc;
        }

        // P -> LDS (C layout -> row-major), then reload as A-frags
        u16* Pw = &Ps[wid * 16 * 64];
#pragma unroll
        for (int t = 0; t < 4; ++t)
#pragma unroll
            for (int r = 0; r < 8; ++r)
                Pw[(r + khalf * 8) * 64 + t * 16 + lrow] = f2bf(pv[t][r]);

        Frag Pa[2];
        {
            const u16* pp = Pw + lrow * 64;
#pragma unroll
            for (int kk = 0; kk < 2; ++kk) {
                Pa[kk].q[0] = *(const v4u*)(pp + kk * 32 + khalf * 8);
                Pa[kk].q[1] = *(const v4u*)(pp + kk * 32 + 16 + khalf * 8);
            }
        }

        // O += P V
#pragma unroll
        for (int kk = 0; kk < 2; ++kk) {
#pragma unroll
            for (int t = 0; t < 4; ++t) {
                Frag bf;
                const int base = (t * 16 + lrow) * 64 + kk * 32 + khalf * 8;
                bf.q[0] = *(const v4u*)&VTs[base];
                bf.q[1] = *(const v4u*)&VTs[base + 16];
                O[t] = wmma_bf16(Pa[kk].v, bf.v, O[t]);
            }
        }
        __syncthreads();
    }

    // normalize + store merged-head bf16 [B,S,E]
    float inv[8];
#pragma unroll
    for (int r = 0; r < 8; ++r) inv[r] = 1.f / fmaxf(lsum[r], 1e-20f);
#pragma unroll
    for (int t = 0; t < 4; ++t)
#pragma unroll
        for (int r = 0; r < 8; ++r) {
            const int row = q0 + r + khalf * 8;
            const int col = h * PD + t * 16 + lrow;
            attn[((size_t)b * PS + row) * PE + col] = f2bf(O[t][r] * inv[r]);
        }
}

// ---------------------------------------------------------------------------
// Host-side launcher
// ---------------------------------------------------------------------------
extern "C" void kernel_launch(void* const* d_in, const int* in_sizes, int n_in,
                              void* d_out, int out_size, void* d_ws, size_t ws_size,
                              hipStream_t stream) {
    const int M = PB * PS;       // 8192
    const int K = PE;            // 1024
    const int N = PE;            // 1024

    const float* x  = (const float*)d_in[0];
    const int* mask = (const int*)d_in[1];
    const float* Wq = (const float*)d_in[2]; const float* bq = (const float*)d_in[3];
    const float* Wk = (const float*)d_in[4]; const float* bk = (const float*)d_in[5];
    const float* Wv = (const float*)d_in[6]; const float* bv = (const float*)d_in[7];
    const float* Wo = (const float*)d_in[8]; const float* bo = (const float*)d_in[9];
    float* out = (float*)d_out;

    char* ws = (char*)d_ws;
    const size_t MiB = 1u << 20;
    u16* xb    = (u16*)(ws + 0);           // 16 MiB
    u16* wqb   = (u16*)(ws + 16 * MiB);    //  2 MiB
    u16* wkb   = (u16*)(ws + 18 * MiB);
    u16* wvb   = (u16*)(ws + 20 * MiB);
    u16* wob   = (u16*)(ws + 22 * MiB);
    u16* qb    = (u16*)(ws + 24 * MiB);    // 16 MiB each, [B*H][S][D]
    u16* kb    = (u16*)(ws + 40 * MiB);
    u16* vb    = (u16*)(ws + 56 * MiB);
    u16* attnb = (u16*)(ws + 72 * MiB);    // 16 MiB, [B][S][E]

    // fp32 -> bf16
    f2bf_kernel<<<1024, 256, 0, stream>>>(x, xb, M * K);
    f2bf_kernel<<<512, 256, 0, stream>>>(Wq, wqb, K * N);
    f2bf_kernel<<<512, 256, 0, stream>>>(Wk, wkb, K * N);
    f2bf_kernel<<<512, 256, 0, stream>>>(Wv, wvb, K * N);
    f2bf_kernel<<<512, 256, 0, stream>>>(Wo, wob, K * N);

    // Q/K/V projections (bf16 WMMA, head-split epilogue)
    dim3 gg(M / 128, N / 128);
    gemm_bf16<true><<<gg, 256, 0, stream>>>(xb, wqb, bq, (void*)qb, M, N, K);
    gemm_bf16<true><<<gg, 256, 0, stream>>>(xb, wkb, bk, (void*)kb, M, N, K);
    gemm_bf16<true><<<gg, 256, 0, stream>>>(xb, wvb, bv, (void*)vb, M, N, K);

    // flash attention
    flash_attn<<<dim3(PB * PH, PS / 128), 256, 0, stream>>>(qb, kb, vb, mask, attnb);

    // output projection (fp32 result)
    gemm_bf16<false><<<gg, 256, 0, stream>>>(attnb, wob, bo, (void*)out, M, N, K);
}
